// LLaDA2MoeExperts_27161373179908
// MI455X (gfx1250) — matrix operand order repaired and checked by reading, not compiled, not run on hardware
//
#include <hip/hip_runtime.h>
#include <hip/hip_bf16.h>

// Problem constants (match reference)
#define T_TOK   8192
#define H_DIM   2048
#define I_DIM   1408
#define E_NUM   16
#define TOPK_N  4
#define S_SLOTS (T_TOK * TOPK_N)   // 32768 routed slots

// GEMM tiling
#define TILE_M 128
#define TILE_N 64
#define TILE_K 32
#define LDSS   40   // padded LDS row stride (elements) to spread banks

typedef __attribute__((ext_vector_type(16))) __bf16 v16bf;
typedef __attribute__((ext_vector_type(8)))  float  v8f;
typedef __attribute__((__vector_size__(4 * sizeof(int)))) int v4i;

union Frag { v16bf v; unsigned u[8]; };

// float -> bf16, round-half-up (2 VALU ops)
__device__ __forceinline__ unsigned short f2bf(float f) {
  return (unsigned short)((__float_as_uint(f) + 0x8000u) >> 16);
}

// Pack two floats into two bf16 (lo in [15:0], hi in [31:16]).
// 2x v_add_nc_u32 + 1x v_perm_b32.
__device__ __forceinline__ unsigned f2bf2(float lo, float hi) {
  unsigned ul = __float_as_uint(lo) + 0x8000u;
  unsigned uh = __float_as_uint(hi) + 0x8000u;
  return __builtin_amdgcn_perm(uh, ul, 0x07060302u);
}

#if __has_builtin(__builtin_amdgcn_global_load_async_to_lds_b128)
#define HAVE_ASYNC_LDS 1
#endif

__device__ __forceinline__ void async_copy16(const void* src, void* dst) {
#ifdef HAVE_ASYNC_LDS
  __builtin_amdgcn_global_load_async_to_lds_b128((v4i*)src, (v4i*)dst, 0, 0);
#else
  uint4 v = *(const uint4*)src;
  unsigned* d = (unsigned*)dst;
  d[0] = v.x; d[1] = v.y; d[2] = v.z; d[3] = v.w;
#endif
}

__device__ __forceinline__ void async_wait() {
#ifdef HAVE_ASYNC_LDS
#if __has_builtin(__builtin_amdgcn_s_wait_asynccnt)
  __builtin_amdgcn_s_wait_asynccnt(0);
#else
  asm volatile("s_wait_asynccnt 0x0" ::: "memory");
#endif
#endif
}

// A fragment: 16x32 bf16 (M x K). Lane L holds row m = L&15.
// Lanes 0-15: V0..3 -> K 0..7 (pairs), V4..7 -> K 16..23.
// Lanes 16-31: V0..3 -> K 8..15,      V4..7 -> K 24..31.
__device__ __forceinline__ v16bf load_a_frag(const unsigned short* As_, int mBase, int lane) {
  Frag f;
  const unsigned short* row = As_ + (mBase + (lane & 15)) * LDSS;
  const int kb = (lane >> 4) * 8;
#pragma unroll
  for (int r = 0; r < 8; ++r) {
    int k = ((r >> 2) * 16) + kb + ((r & 3) * 2);
    f.u[r] = *(const unsigned*)(row + k);
  }
  return f.v;
}

// B fragment: 32x16 bf16 (K x N). Lane L holds col n = L&15.
// Lanes 0-15: V0..3 -> K 0..7, V4..7 -> K 16..23; lanes 16-31: K 8..15, K 24..31.
__device__ __forceinline__ v16bf load_b_frag(const unsigned short* Bs_, int nBase, int lane) {
  Frag f;
  const unsigned short* row = Bs_ + (nBase + (lane & 15)) * LDSS + (lane >> 4) * 8;
#pragma unroll
  for (int r = 0; r < 8; ++r) {
    f.u[r] = *(const unsigned*)(row + ((r >> 2) * 16) + ((r & 3) * 2));
  }
  return f.v;
}

__device__ __forceinline__ v8f wmma_bf16(v16bf a, v16bf b, v8f c) {
  return __builtin_amdgcn_wmma_f32_16x16x32_bf16(false, a, false, b, (short)0, c, false, false);
}

// ---------------- routing / setup kernels ----------------

__global__ void zero_out_kernel(float4* __restrict__ p) {
  p[(size_t)blockIdx.x * blockDim.x + threadIdx.x] = make_float4(0.f, 0.f, 0.f, 0.f);
}

__global__ void init_counts_kernel(int* __restrict__ counts) {
  if (threadIdx.x < E_NUM) counts[threadIdx.x] = 0;
}

__global__ void count_kernel(const int* __restrict__ sel, int* __restrict__ counts) {
  int i = blockIdx.x * blockDim.x + threadIdx.x;
  if (i < S_SLOTS) atomicAdd(&counts[sel[i]], 1);
}

__global__ void scan_kernel(const int* __restrict__ counts, int* __restrict__ offs,
                            int* __restrict__ cursor) {
  if (threadIdx.x == 0) {
    int acc = 0;
    for (int e = 0; e < E_NUM; ++e) {
      offs[e] = acc;
      cursor[e] = acc;
      acc += counts[e];
    }
    offs[E_NUM] = acc;
  }
}

__global__ void scatter_kernel(const int* __restrict__ sel, const float* __restrict__ rw,
                               int* __restrict__ cursor, int* __restrict__ tok_id,
                               float* __restrict__ tok_w) {
  int i = blockIdx.x * blockDim.x + threadIdx.x;
  if (i < S_SLOTS) {
    int e = sel[i];
    int p = atomicAdd(&cursor[e], 1);
    tok_id[p] = i / TOPK_N;
    tok_w[p]  = rw[i];
  }
}

// ---------------- gate/up GEMM + SwiGLU ----------------
// grid: (I/64, E, 16). block tile 128(M slots) x 64(N of I), K over H.
// Double-buffered LDS + register-pipelined staging: loads for tile k+1 are
// issued before the WMMAs of tile k; their wait lands at tile k+1's convert.
__global__ __launch_bounds__(256) void moe_gateup_kernel(
    const float* __restrict__ x, const float* __restrict__ gate,
    const float* __restrict__ up, const int* __restrict__ offs,
    const int* __restrict__ tok_id, unsigned short* __restrict__ act) {
  __shared__ unsigned short As[2][TILE_M * LDSS];
  __shared__ unsigned short BgS[2][TILE_N * LDSS];
  __shared__ unsigned short BuS[2][TILE_N * LDSS];

  const int tid  = threadIdx.x;
  const int lane = tid & 31;
  const int w    = tid >> 5;
  const int wm   = (w >> 1) * 32;   // 0,32,64,96
  const int wn   = (w & 1) * 32;    // 0,32
  const int e    = blockIdx.y;
  const int n0   = blockIdx.x * TILE_N;
  const int seg0 = offs[e];
  const int ne   = offs[e + 1] - seg0;
  const size_t wbase = (size_t)e * I_DIM * H_DIM;

  const v8f zacc = {0.f, 0.f, 0.f, 0.f, 0.f, 0.f, 0.f, 0.f};
  const int rBase = tid >> 3;        // 0..31
  const int cOff  = (tid & 7) << 2;  // 0..28

  // loop-invariant B-row pointers (this thread's 2 gate rows + 2 up rows)
  const float* gp[2];
  const float* qp[2];
#pragma unroll
  for (int i = 0; i < 2; ++i) {
    size_t roff = wbase + (size_t)(n0 + rBase + i * 32) * H_DIM + cOff;
    gp[i] = gate + roff;
    qp[i] = up + roff;
  }

  for (int mt = blockIdx.z; mt * TILE_M < ne; mt += gridDim.z) {
    const int m0 = mt * TILE_M;

    // cache this thread's 4 gathered A-row pointers in registers
    const float* ap[4];
#pragma unroll
    for (int i = 0; i < 4; ++i) {
      int row = m0 + rBase + i * 32;
      ap[i] = (row < ne) ? (x + (size_t)tok_id[seg0 + row] * H_DIM + cOff) : (const float*)0;
    }

    v8f cg[2][2], cu[2][2];
#pragma unroll
    for (int i = 0; i < 2; ++i)
#pragma unroll
      for (int j = 0; j < 2; ++j) { cg[i][j] = zacc; cu[i][j] = zacc; }

    // staging registers (one tile in flight)
    float4 rA[4], rG[2], rU[2];

    auto gload = [&](int kk) {
#pragma unroll
      for (int i = 0; i < 4; ++i) {
        rA[i] = make_float4(0.f, 0.f, 0.f, 0.f);
        if (ap[i]) rA[i] = *(const float4*)(ap[i] + kk);
      }
#pragma unroll
      for (int i = 0; i < 2; ++i) {
        rG[i] = *(const float4*)(gp[i] + kk);
        rU[i] = *(const float4*)(qp[i] + kk);
      }
    };

    auto store_lds = [&](int buf) {
#pragma unroll
      for (int i = 0; i < 4; ++i) {
        unsigned* d = (unsigned*)(As[buf] + (rBase + i * 32) * LDSS + cOff);
        d[0] = f2bf2(rA[i].x, rA[i].y);
        d[1] = f2bf2(rA[i].z, rA[i].w);
      }
#pragma unroll
      for (int i = 0; i < 2; ++i) {
        unsigned* dg = (unsigned*)(BgS[buf] + (rBase + i * 32) * LDSS + cOff);
        dg[0] = f2bf2(rG[i].x, rG[i].y);
        dg[1] = f2bf2(rG[i].z, rG[i].w);
        unsigned* du = (unsigned*)(BuS[buf] + (rBase + i * 32) * LDSS + cOff);
        du[0] = f2bf2(rU[i].x, rU[i].y);
        du[1] = f2bf2(rU[i].z, rU[i].w);
      }
    };

    gload(0);
    const int NK = H_DIM / TILE_K;   // 64, even
    for (int ks = 0; ks < NK; ++ks) {
      const int cur = ks & 1;
      store_lds(cur);                       // converts tile ks (waits its loads here)
      __syncthreads();                      // publish buffer `cur`; `cur^1` free
      if (ks + 1 < NK) gload((ks + 1) * TILE_K);   // issue next tile's loads early

      const unsigned short* Ab = As[cur];
      const unsigned short* Gb = BgS[cur];
      const unsigned short* Ub = BuS[cur];
      v16bf a0  = load_a_frag(Ab, wm, lane);
      v16bf a1  = load_a_frag(Ab, wm + 16, lane);
      v16bf bg0 = load_b_frag(Gb, wn, lane);
      v16bf bg1 = load_b_frag(Gb, wn + 16, lane);
      v16bf bu0 = load_b_frag(Ub, wn, lane);
      v16bf bu1 = load_b_frag(Ub, wn + 16, lane);

      cg[0][0] = wmma_bf16(a0, bg0, cg[0][0]);
      cg[0][1] = wmma_bf16(a0, bg1, cg[0][1]);
      cg[1][0] = wmma_bf16(a1, bg0, cg[1][0]);
      cg[1][1] = wmma_bf16(a1, bg1, cg[1][1]);
      cu[0][0] = wmma_bf16(a0, bu0, cu[0][0]);
      cu[0][1] = wmma_bf16(a0, bu1, cu[0][1]);
      cu[1][0] = wmma_bf16(a1, bu0, cu[1][0]);
      cu[1][1] = wmma_bf16(a1, bu1, cu[1][1]);
    }

    // SwiGLU epilogue -> bf16 activations in workspace
#pragma unroll
    for (int i = 0; i < 2; ++i)
#pragma unroll
      for (int j = 0; j < 2; ++j) {
        int col = n0 + wn + j * 16 + (lane & 15);
#pragma unroll
        for (int r = 0; r < 8; ++r) {
          int mr = wm + i * 16 + ((lane >> 4) * 8) + r;
          if (m0 + mr < ne) {
            float gv = cg[i][j][r];
            float uv = cu[i][j][r];
            float s  = gv / (1.f + __expf(-gv));
            act[(size_t)(seg0 + m0 + mr) * I_DIM + col] = f2bf(s * uv);
          }
        }
      }
  }
}

// ---------------- down GEMM + weighted atomic combine ----------------
// grid: (H/64, E, 16). block tile 128(M slots) x 64(N of H), K over I.
// A operand is already bf16 -> async copy global->LDS (ASYNCcnt path),
// issued one iteration ahead (after the barrier) so it overlaps the WMMAs.
__global__ __launch_bounds__(256) void moe_down_kernel(
    const float* __restrict__ down, const unsigned short* __restrict__ act,
    const int* __restrict__ offs, const int* __restrict__ tok_id,
    const float* __restrict__ tok_w, float* __restrict__ out) {
  __shared__ unsigned short As[2][TILE_M * LDSS];
  __shared__ unsigned short BdS[2][TILE_N * LDSS];

  const int tid  = threadIdx.x;
  const int lane = tid & 31;
  const int w    = tid >> 5;
  const int wm   = (w >> 1) * 32;
  const int wn   = (w & 1) * 32;
  const int e    = blockIdx.y;
  const int n0   = blockIdx.x * TILE_N;
  const int seg0 = offs[e];
  const int ne   = offs[e + 1] - seg0;
  const size_t wbase = (size_t)e * H_DIM * I_DIM;

  const v8f zacc = {0.f, 0.f, 0.f, 0.f, 0.f, 0.f, 0.f, 0.f};
  const int rBaseA = tid >> 2;        // 0..63
  const int c8     = (tid & 3) << 3;  // 0,8,16,24 (bf16 elems)
  const int rBaseB = tid >> 3;        // 0..31
  const int cOff   = (tid & 7) << 2;  // 0..28

  // loop-invariant B-row pointers
  const float* dp[2];
#pragma unroll
  for (int i = 0; i < 2; ++i)
    dp[i] = down + wbase + (size_t)(n0 + rBaseB + i * 32) * I_DIM + cOff;

  for (int mt = blockIdx.z; mt * TILE_M < ne; mt += gridDim.z) {
    const int m0 = mt * TILE_M;

    // this thread's 2 activation-row pointers
    const unsigned short* arp[2];
#pragma unroll
    for (int i = 0; i < 2; ++i) {
      int row = m0 + rBaseA + i * 64;
      arp[i] = (row < ne) ? (act + (size_t)(seg0 + row) * I_DIM + c8) : (const unsigned short*)0;
    }

    v8f cc[2][2];
#pragma unroll
    for (int i = 0; i < 2; ++i)
#pragma unroll
      for (int j = 0; j < 2; ++j) cc[i][j] = zacc;

    float4 rB[2];

    auto gloadB = [&](int kk) {
#pragma unroll
      for (int i = 0; i < 2; ++i) rB[i] = *(const float4*)(dp[i] + kk);
    };

    auto storeB = [&](int buf) {
#pragma unroll
      for (int i = 0; i < 2; ++i) {
        unsigned* d = (unsigned*)(BdS[buf] + (rBaseB + i * 32) * LDSS + cOff);
        d[0] = f2bf2(rB[i].x, rB[i].y);
        d[1] = f2bf2(rB[i].z, rB[i].w);
      }
    };

    auto asyncA = [&](int buf, int kk) {
#pragma unroll
      for (int i = 0; i < 2; ++i) {
        void* dst = (void*)(As[buf] + (rBaseA + i * 64) * LDSS + c8);
        if (arp[i]) {
          async_copy16(arp[i] + kk, dst);
        } else {
          unsigned* d = (unsigned*)dst;
          d[0] = 0u; d[1] = 0u; d[2] = 0u; d[3] = 0u;
        }
      }
    };

    gloadB(0);
    asyncA(0, 0);
    const int NK = I_DIM / TILE_K;   // 44, even
    for (int ks = 0; ks < NK; ++ks) {
      const int cur = ks & 1;
      storeB(cur);                          // converts B tile ks (waits its loads here)
      async_wait();                         // A tile ks async copies landed
      __syncthreads();                      // publish buffers `cur`; `cur^1` free
      if (ks + 1 < NK) {
        gloadB((ks + 1) * TILE_K);          // issue next B loads early
        asyncA(cur ^ 1, (ks + 1) * TILE_K); // issue next A async copies (post-barrier)
      }

      const unsigned short* Ab = As[cur];
      const unsigned short* Bb = BdS[cur];
      v16bf a0 = load_a_frag(Ab, wm, lane);
      v16bf a1 = load_a_frag(Ab, wm + 16, lane);
      v16bf b0 = load_b_frag(Bb, wn, lane);
      v16bf b1 = load_b_frag(Bb, wn + 16, lane);

      cc[0][0] = wmma_bf16(a0, b0, cc[0][0]);
      cc[0][1] = wmma_bf16(a0, b1, cc[0][1]);
      cc[1][0] = wmma_bf16(a1, b0, cc[1][0]);
      cc[1][1] = wmma_bf16(a1, b1, cc[1][1]);
    }

    // combine: out[token, h] += routing_w * y
#pragma unroll
    for (int i = 0; i < 2; ++i)
#pragma unroll
      for (int j = 0; j < 2; ++j) {
        int col = n0 + wn + j * 16 + (lane & 15);
#pragma unroll
        for (int r = 0; r < 8; ++r) {
          int mr = wm + i * 16 + ((lane >> 4) * 8) + r;
          if (m0 + mr < ne) {
            int slot = seg0 + m0 + mr;
            float wv = tok_w[slot];
            int t = tok_id[slot];
            unsafeAtomicAdd(out + (size_t)t * H_DIM + col, cc[i][j][r] * wv);
          }
        }
      }
  }
}

// ---------------- launcher ----------------

extern "C" void kernel_launch(void* const* d_in, const int* in_sizes, int n_in,
                              void* d_out, int out_size, void* d_ws, size_t ws_size,
                              hipStream_t stream) {
  (void)in_sizes; (void)n_in; (void)out_size; (void)ws_size;

  const float* x    = (const float*)d_in[0];   // [T,H]
  const float* rw   = (const float*)d_in[1];   // [T,TOPK]
  const int*   sel  = (const int*)d_in[2];     // [T,TOPK]
  const float* gate = (const float*)d_in[3];   // [E,I,H]
  const float* up   = (const float*)d_in[4];   // [E,I,H]
  const float* down = (const float*)d_in[5];   // [E,H,I]
  float* out = (float*)d_out;                  // [T,H]

  // workspace layout
  int*   counts = (int*)d_ws;                         // [16]
  int*   offs   = counts + 16;                        // [17]
  int*   cursor = counts + 64;                        // [16]
  int*   tok_id = counts + 128;                       // [S]
  float* tok_w  = (float*)(counts + 128 + S_SLOTS);   // [S]
  unsigned short* act =
      (unsigned short*)((char*)d_ws + 512 + (size_t)S_SLOTS * 8);  // bf16 [S, I]

  // 1) zero output + counters
  zero_out_kernel<<<(T_TOK * H_DIM) / 1024, 256, 0, stream>>>((float4*)out);
  init_counts_kernel<<<1, 64, 0, stream>>>(counts);

  // 2) routing: histogram -> scan -> scatter
  count_kernel<<<S_SLOTS / 256, 256, 0, stream>>>(sel, counts);
  scan_kernel<<<1, 32, 0, stream>>>(counts, offs, cursor);
  scatter_kernel<<<S_SLOTS / 256, 256, 0, stream>>>(sel, rw, cursor, tok_id, tok_w);

  // 3) gate/up GEMM + SwiGLU  (I/64 = 22 N-tiles)
  moe_gateup_kernel<<<dim3(I_DIM / TILE_N, E_NUM, 16), 256, 0, stream>>>(
      x, gate, up, offs, tok_id, act);

  // 4) down GEMM + weighted combine  (H/64 = 32 N-tiles)
  moe_down_kernel<<<dim3(H_DIM / TILE_N, E_NUM, 16), 256, 0, stream>>>(
      down, act, offs, tok_id, tok_w, out);
}
